// FEM_best_55439437857362
// MI455X (gfx1250) — matrix operand order repaired and checked by reading, not compiled, not run on hardware
//
#include <hip/hip_runtime.h>

// ---------------------------------------------------------------------------
// CDNA5 (gfx1250) wave32 WMMA bf16 pipeline for the FEM cross-attention block.
// Round 4: templated conv (KSZ/PAD/H/W compile-time -> no runtime int division)
// and templated attention layout modes (dead-branch elimination). WMMA hot
// path unchanged: ds_load_b128 fragments -> v_wmma_f32_16x16x32_bf16.
// ---------------------------------------------------------------------------

typedef __attribute__((ext_vector_type(16))) __bf16 v16bf;
typedef __attribute__((ext_vector_type(8)))  __bf16 v8bf;
typedef __attribute__((ext_vector_type(8)))  float  v8f;
typedef __attribute__((ext_vector_type(4)))  float  v4f;
typedef __attribute__((ext_vector_type(4)))  int    v4i;

#if defined(__has_builtin)
#  if __has_builtin(__builtin_amdgcn_global_load_async_to_lds_b128)
#    define FEM_ASYNC_LDS 1
#  endif
#endif

__device__ __forceinline__ __bf16 f2bf(float f) {
  union { float f; unsigned u; } in; in.f = f;
  unsigned r = in.u + 0x7FFFu + ((in.u >> 16) & 1u);   // round-to-nearest-even
  union { unsigned short u; __bf16 b; } o;
  o.u = (unsigned short)(r >> 16);
  return o.b;
}

__device__ __forceinline__ v8bf loadu8(const __bf16* p) {   // unaligned-safe
  v8bf v;
  __builtin_memcpy(&v, p, sizeof(v));
  return v;
}

__device__ __forceinline__ v8f wmma_bf16(const v16bf a, const v16bf b, const v8f c) {
  return __builtin_amdgcn_wmma_f32_16x16x32_bf16(false, a, false, b, (short)0, c,
                                                 false, false);
}

// A fragment from row-major tile (element (m,k) = base[m*rs + k], k contiguous).
__device__ __forceinline__ v16bf load_frag_a_vec(const __bf16* base, int rs) {
  int lane = threadIdx.x & 31;
  int m  = lane & 15;
  int kh = (lane >> 4) & 1;
  const __bf16* p = base + m * rs + kh * 8;
  v8bf lo = *(const v8bf*)p;          // K block 0..7 / 8..15
  v8bf hi = *(const v8bf*)(p + 16);   // K block 16..23 / 24..31
  return __builtin_shufflevector(lo, hi, 0, 1, 2, 3, 4, 5, 6, 7,
                                 8, 9, 10, 11, 12, 13, 14, 15);
}

// B fragment from TRANSPOSED tile (element (k,n) = base[n*rs + k], k contiguous).
__device__ __forceinline__ v16bf load_frag_bT(const __bf16* base, int rs) {
  int lane = threadIdx.x & 31;
  int n  = lane & 15;
  int kb = (lane >> 4) ? 8 : 0;
  const __bf16* p = base + n * rs + kb;
  v8bf lo = *(const v8bf*)p;
  v8bf hi = *(const v8bf*)(p + 16);
  return __builtin_shufflevector(lo, hi, 0, 1, 2, 3, 4, 5, 6, 7,
                                 8, 9, 10, 11, 12, 13, 14, 15);
}

__device__ __forceinline__ void pack_store2(__bf16* dst, __bf16 a, __bf16 b) {
  union { __bf16 h[2]; unsigned u; } pk;
  pk.h[0] = a; pk.h[1] = b;
  *(unsigned*)dst = pk.u;            // dst 4B-aligned by construction
}

// ---------------------------------------------------------------------------
__global__ void cvt_f32_bf16_kernel(const float* __restrict__ in,
                                    __bf16* __restrict__ out, int n) {
  int i = blockIdx.x * blockDim.x + threadIdx.x;
  if (i < n) out[i] = f2bf(in[i]);
}

// ---------------------------------------------------------------------------
// 8 consecutive spatial positions of one im2col row k. KSZ/PAD/H/W are
// compile-time: all divisions fold to shifts / mul-shift sequences.
// ---------------------------------------------------------------------------
template <int KSZ, int PAD, int HH, int WW>
__device__ __forceinline__ v8bf im2col8(const __bf16* __restrict__ Xb, int k,
                                        int nbase) {
  const int N = HH * WW;
  if (KSZ == 1) return *(const v8bf*)(Xb + (long)k * N + nbase);
  const int kk2 = KSZ * KSZ;
  int ci  = k / kk2;
  int rem = k - ci * kk2;
  int r   = rem / KSZ;
  int s2  = rem - r * KSZ;
  int h   = nbase / WW;
  int w0  = nbase - h * WW;
  int ih  = h + r - PAD;
  v8bf out;
  if (ih < 0 || ih >= HH) {
#pragma unroll
    for (int j = 0; j < 8; ++j) out[j] = f2bf(0.f);
    return out;
  }
  const __bf16* rowp = Xb + (long)ci * N + (long)ih * WW;
  int iw0 = w0 + s2 - PAD;
  if (iw0 >= 0 && iw0 + 8 <= WW) return loadu8(rowp + iw0);
#pragma unroll
  for (int j = 0; j < 8; ++j) {
    int iw = iw0 + j;
    out[j] = (iw >= 0 && iw < WW) ? rowp[iw] : f2bf(0.f);
  }
  return out;
}

// ---------------------------------------------------------------------------
// Implicit-GEMM conv, bf16 WMMA, f32 accumulate, fused epilogue.
// Block: 256 threads = 8 waves. M-tile = 16 out-channels, N-tile = 128 pixels.
// ldsA row-major [m][k]; ldsBt transposed [n][k] so frags are ds_load_b128.
// ---------------------------------------------------------------------------
template <int KSZ, int PAD, int HH, int WW>
__global__ __launch_bounds__(256) void conv_gemm_kernel(
    const __bf16* __restrict__ X, long x_bs,
    const __bf16* __restrict__ Wt,
    const float* __restrict__ bias,
    const float* __restrict__ bn_g, const float* __restrict__ bn_b,
    const float* __restrict__ bn_m, const float* __restrict__ bn_v,
    float*  __restrict__ outF, long of_bs, int of_cs, int of_ns,
    __bf16* __restrict__ outB, long ob_bs, int ob_cs, int ob_ns,
    int Cout, int Cin, int relu, int negate) {
  const int Ktot = Cin * KSZ * KSZ;
  const int b    = blockIdx.z;
  const int m0   = blockIdx.x * 16;
  const int n0   = blockIdx.y * 128;
  const int tid  = threadIdx.x;
  const int wid  = tid >> 5;
  const int lane = tid & 31;

  __shared__ __attribute__((aligned(16))) __bf16 ldsA[16 * 32];    // [m][k]
  __shared__ __attribute__((aligned(16))) __bf16 ldsBt[128 * 32];  // [n][k]

  const __bf16* Xb = X + (long)b * x_bs;
  v8f acc = {0.f, 0.f, 0.f, 0.f, 0.f, 0.f, 0.f, 0.f};

  const int kpair = (tid >> 4) * 2;      // 0..30
  const int npos  = (tid & 15) * 8;      // 0..120

  for (int k0 = 0; k0 < Ktot; k0 += 32) {
    // ---- A tile: 16x32 weights, b128 in / b128 out ----
    if (tid < 64) {
      int m  = tid >> 2;
      int kg = (tid & 3) * 8;
      *(v8bf*)&ldsA[m * 32 + kg] =
          *(const v8bf*)&Wt[(long)(m0 + m) * Ktot + k0 + kg];
    }
    if (k0 + 32 < Ktot)
      __builtin_prefetch(&Wt[(long)m0 * Ktot + k0 + 32], 0, 1);

    // ---- B tile: im2col 32x128, transposed store as packed b32 ----
    v8bf r0 = im2col8<KSZ, PAD, HH, WW>(Xb, k0 + kpair,     n0 + npos);
    v8bf r1 = im2col8<KSZ, PAD, HH, WW>(Xb, k0 + kpair + 1, n0 + npos);
#pragma unroll
    for (int j = 0; j < 8; ++j)
      pack_store2(&ldsBt[(npos + j) * 32 + kpair], r0[j], r1[j]);

    __syncthreads();
    v16bf af = load_frag_a_vec(ldsA, 32);
    v16bf bf = load_frag_bT(&ldsBt[(wid * 16) * 32], 32);
    acc = wmma_bf16(af, bf, acc);
    __syncthreads();
  }

  // ---- fused epilogue (C layout: lane -> col n; VGPR r -> row r (+8)) ----
  const int nl   = lane & 15;
  const int mofs = (lane >> 4) ? 8 : 0;
  const int n    = n0 + wid * 16 + nl;
#pragma unroll
  for (int r = 0; r < 8; ++r) {
    int c = m0 + r + mofs;
    float v = acc[r];
    if (bias) v += bias[c];
    if (bn_g) {
      float sc = bn_g[c] * rsqrtf(bn_v[c] + 1e-5f);
      v = (v - bn_m[c]) * sc + bn_b[c];
    }
    if (relu)   v = fmaxf(v, 0.f);
    if (negate) v = -v;
    if (outF) outF[(long)b * of_bs + (long)c * of_cs + (long)n * of_ns] = v;
    if (outB) outB[(long)b * ob_bs + (long)c * ob_cs + (long)n * ob_ns] = f2bf(v);
  }
}

// ---------------------------------------------------------------------------
// Flash-style abs-score attention: Out[c,n] = sum_m softmax_m(|K[n,:]Q[:,m]|) V[m,c]
// A_CC: A operand is channel-contiguous (a_cs==1). B_CC: B operand is
// channel-contiguous (b_cs==1). Untaken paths are dead-code-eliminated.
// ---------------------------------------------------------------------------
template <bool A_CC, bool B_CC>
__global__ __launch_bounds__(256) void attn_kernel(
    const __bf16* __restrict__ Aop, int a_ns, int a_cs, long a_bs,  // (n,c)
    const __bf16* __restrict__ Bop, int b_cs, int b_ms, long b_bs,  // (c,m)
    const __bf16* __restrict__ Vop, long v_bs,                      // [N,C] rowmaj
    float* __restrict__ Out, long o_bs,                             // [C,N]
    int N, int C) {                                                 // C == 128
  const int b    = blockIdx.y;
  const int n0   = blockIdx.x * 64;
  const int tid  = threadIdx.x;
  const int wid  = tid >> 5;
  const int lane = tid & 31;
  const int nl   = lane & 15;
  const int mofs = (lane >> 4) ? 8 : 0;

  __shared__ __attribute__((aligned(16))) __bf16 ldsK [64 * 128];  // [n][c]
  __shared__ __attribute__((aligned(16))) __bf16 ldsQt[64 * 128];  // [m][c]
  __shared__ __attribute__((aligned(16))) __bf16 ldsVt[128 * 64];  // [c][m]
  __shared__ __attribute__((aligned(16))) float  ldsS [64 * 64];   // scores
  __shared__ __attribute__((aligned(16))) __bf16 ldsP [64 * 64];   // probs
  __shared__ float red[256];
  __shared__ float row_max[64], row_sum[64], row_scale[64];

  const __bf16* Ab = Aop + (long)b * a_bs;
  const __bf16* Bb = Bop + (long)b * b_bs;
  const __bf16* Vb = Vop + (long)b * v_bs;

  // ---- load the 64 A-rows once ----
  if (A_CC) {                            // contiguous in c
#ifdef FEM_ASYNC_LDS
    for (int u = tid; u < 1024; u += 256) {
      int n = u >> 4, cg = (u & 15) * 8;
      __builtin_amdgcn_global_load_async_to_lds_b128(
          (v4i*)&Ab[(long)(n0 + n) * a_ns + cg],
          (v4i*)&ldsK[n * 128 + cg], 0, 0);
    }
    __builtin_amdgcn_s_wait_asynccnt(0);
#else
    for (int u = tid; u < 1024; u += 256) {
      int n = u >> 4, cg = (u & 15) * 8;
      *(v8bf*)&ldsK[n * 128 + cg] = *(const v8bf*)&Ab[(long)(n0 + n) * a_ns + cg];
    }
#endif
  } else {                               // a_ns == 1: contiguous in n, transpose
    for (int u = tid; u < 512; u += 256) {
      int ng = u & 7, cp = u >> 3;
      int n8 = ng * 8, c2 = cp * 2;
      v8bf ra = *(const v8bf*)&Ab[(long)c2 * a_cs + (n0 + n8)];
      v8bf rb = *(const v8bf*)&Ab[(long)(c2 + 1) * a_cs + (n0 + n8)];
#pragma unroll
      for (int j = 0; j < 8; ++j)
        pack_store2(&ldsK[(n8 + j) * 128 + c2], ra[j], rb[j]);
    }
  }
  if (tid < 64) { row_max[tid] = -1e30f; row_sum[tid] = 0.f; }

  v8f acc[4];
#pragma unroll
  for (int i = 0; i < 4; ++i)
    acc[i] = (v8f){0.f, 0.f, 0.f, 0.f, 0.f, 0.f, 0.f, 0.f};

  const int srow = tid >> 2;             // softmax: 4 threads per row
  const int sq   = tid & 3;

  for (int m0 = 0; m0 < N; m0 += 64) {
    // ---- B-operand tile -> ldsQt[m][c] ----
    if (B_CC) {                          // element (c,m)=Bb[m*b_ms+c]: row copy
#ifdef FEM_ASYNC_LDS
      for (int u = tid; u < 1024; u += 256) {
        int mm = u >> 4, cg = (u & 15) * 8;
        __builtin_amdgcn_global_load_async_to_lds_b128(
            (v4i*)&Bb[(long)(m0 + mm) * b_ms + cg],
            (v4i*)&ldsQt[mm * 128 + cg], 0, 0);
      }
      __builtin_amdgcn_s_wait_asynccnt(0);
#else
      for (int u = tid; u < 1024; u += 256) {
        int mm = u >> 4, cg = (u & 15) * 8;
        *(v8bf*)&ldsQt[mm * 128 + cg] =
            *(const v8bf*)&Bb[(long)(m0 + mm) * b_ms + cg];
      }
#endif
    } else {                             // b_ms == 1: contiguous m, transpose
      for (int u = tid; u < 512; u += 256) {
        int mg = u & 7, cp = u >> 3;
        int m8 = mg * 8, c2 = cp * 2;
        v8bf ra = *(const v8bf*)&Bb[(long)c2 * b_cs + (m0 + m8)];
        v8bf rb = *(const v8bf*)&Bb[(long)(c2 + 1) * b_cs + (m0 + m8)];
#pragma unroll
        for (int j = 0; j < 8; ++j)
          pack_store2(&ldsQt[(m8 + j) * 128 + c2], ra[j], rb[j]);
      }
    }
    // ---- V tile -> ldsVt[c][m] (transpose of [m][c]) ----
    for (int u = tid; u < 512; u += 256) {
      int cg = u & 15, mp = u >> 4;
      int c8 = cg * 8, m2 = mp * 2;
      v8bf ra = *(const v8bf*)&Vb[(long)(m0 + m2) * C + c8];
      v8bf rb = *(const v8bf*)&Vb[(long)(m0 + m2 + 1) * C + c8];
#pragma unroll
      for (int j = 0; j < 8; ++j)
        pack_store2(&ldsVt[(c8 + j) * 64 + m2], ra[j], rb[j]);
    }
    __syncthreads();

    // ---- scores: 16 sub-tiles of 16x16, 2 per wave, K over C=128 ----
#pragma unroll
    for (int t = 0; t < 2; ++t) {
      int st = wid + t * 8;
      int ni = st >> 2, mi = st & 3;
      v8f s = (v8f){0.f, 0.f, 0.f, 0.f, 0.f, 0.f, 0.f, 0.f};
#pragma unroll
      for (int kk = 0; kk < 4; ++kk) {
        v16bf af = load_frag_a_vec(&ldsK [ni * 16 * 128 + kk * 32], 128);
        v16bf bq = load_frag_bT  (&ldsQt[mi * 16 * 128 + kk * 32], 128);
        s = wmma_bf16(af, bq, s);
      }
#pragma unroll
      for (int r = 0; r < 8; ++r)
        ldsS[(ni * 16 + r + mofs) * 64 + mi * 16 + nl] = fabsf(s[r]);
    }
    __syncthreads();

    // ---- parallel online softmax: 4 threads per row, float4 reads ----
    const float* sr = &ldsS[srow * 64 + sq * 16];
    float lm = 0.f;                      // |scores| >= 0
#pragma unroll
    for (int jv = 0; jv < 4; ++jv) {
      v4f x = *(const v4f*)(sr + jv * 4);
      lm = fmaxf(lm, fmaxf(fmaxf(x[0], x[1]), fmaxf(x[2], x[3])));
    }
    red[tid] = lm;
    __syncthreads();
    if (sq == 0) {
      float tmax = fmaxf(fmaxf(red[tid], red[tid + 1]),
                         fmaxf(red[tid + 2], red[tid + 3]));
      float mx = row_max[srow];
      float Mn = fmaxf(mx, tmax);
      row_scale[srow] = __expf(mx - Mn);
      row_max[srow]   = Mn;
    }
    __syncthreads();
    {
      float Mn = row_max[srow];
      float ls = 0.f;
      v8bf p0, p1;
#pragma unroll
      for (int j = 0; j < 8; ++j) {
        float p = __expf(sr[j] - Mn);
        ls += p; p0[j] = f2bf(p);
      }
#pragma unroll
      for (int j = 0; j < 8; ++j) {
        float p = __expf(sr[8 + j] - Mn);
        ls += p; p1[j] = f2bf(p);
      }
      __bf16* pr = &ldsP[srow * 64 + sq * 16];
      *(v8bf*)pr = p0;
      *(v8bf*)(pr + 8) = p1;
      red[tid] = ls;
    }
    __syncthreads();
    if (sq == 0)
      row_sum[srow] = row_sum[srow] * row_scale[srow] +
                      red[tid] + red[tid + 1] + red[tid + 2] + red[tid + 3];

    // ---- rescale accumulators, then P @ V ----
#pragma unroll
    for (int ni = 0; ni < 4; ++ni) {
#pragma unroll
      for (int r = 0; r < 8; ++r)
        acc[ni][r] *= row_scale[ni * 16 + r + mofs];
#pragma unroll
      for (int kk = 0; kk < 2; ++kk) {
        v16bf af = load_frag_a_vec(&ldsP[ni * 16 * 64 + kk * 32], 64);
        v16bf bv = load_frag_bT  (&ldsVt[(wid * 16) * 64 + kk * 32], 64);
        acc[ni] = wmma_bf16(af, bv, acc[ni]);
      }
    }
    __syncthreads();
  }

  // ---- normalize and store Out[c*N + n] ----
  const int c = wid * 16 + nl;
  float* Ob = Out + (long)b * o_bs;
#pragma unroll
  for (int ni = 0; ni < 4; ++ni) {
#pragma unroll
    for (int r = 0; r < 8; ++r) {
      int row = ni * 16 + r + mofs;
      Ob[(long)c * N + (n0 + row)] = acc[ni][r] / row_sum[row];
    }
  }
}

// ---------------------------------------------------------------------------
__global__ void pool_kernel(const float* __restrict__ in, float* __restrict__ out,
                            int N) {
  __shared__ float red[256];
  const float* p = in + (long)blockIdx.x * N;
  float s = 0.f;
  for (int i = threadIdx.x; i < N; i += 256) s += p[i];
  red[threadIdx.x] = s;
  __syncthreads();
  for (int off = 128; off > 0; off >>= 1) {
    if ((int)threadIdx.x < off) red[threadIdx.x] += red[threadIdx.x + off];
    __syncthreads();
  }
  if (threadIdx.x == 0) out[blockIdx.x] = red[0] / (float)N;
}

__global__ void gate_kernel(const float* __restrict__ pooled,
                            const float* __restrict__ w1, const float* __restrict__ b1,
                            const float* __restrict__ w2, const float* __restrict__ b2,
                            float* __restrict__ gate, int B, int C, int Cr) {
  __shared__ float h[64];
  int tid = threadIdx.x;
  if (tid < B * Cr) {
    int b = tid / Cr, j = tid % Cr;
    float s = b1[j];
    for (int c = 0; c < C; ++c) s += pooled[b * C + c] * w1[j * C + c];
    h[tid] = fmaxf(s, 0.f);
  }
  __syncthreads();
  if (tid < B * C) {
    int b = tid / C, c = tid % C;
    float s = b2[c];
    for (int j = 0; j < Cr; ++j) s += h[b * Cr + j] * w2[c * Cr + j];
    gate[tid] = 1.f / (1.f + __expf(-s));
  }
}

__global__ void fuse_kernel(const float* __restrict__ p,
                            const float* __restrict__ stem,
                            const float* __restrict__ gate,
                            float* __restrict__ outF, __bf16* __restrict__ outB,
                            int C, int N, int chofs, int Ctot, long total) {
  long idx = (long)blockIdx.x * blockDim.x + threadIdx.x;
  if (idx >= total) return;
  int n = (int)(idx % N);
  long t = idx / N;
  int c = (int)(t % C);
  int b = (int)(t / C);
  float e = gate[b * C + c] * p[idx] + stem[idx];
  outF[idx] = e;
  outB[(long)b * Ctot * N + (long)(chofs + c) * N + n] = f2bf(e);
}

// ---------------------------------------------------------------------------
extern "C" void kernel_launch(void* const* d_in, const int* in_sizes, int n_in,
                              void* d_out, int out_size, void* d_ws, size_t ws_size,
                              hipStream_t stream) {
  (void)in_sizes; (void)n_in; (void)out_size; (void)ws_size;
  const int B = 2, Cin = 256, Ci = 128, Cr = 8;
  const int N = 64 * 64;                     // 4096
  const long CN  = (long)Ci * N;             // 524288
  const long BCN = (long)B * CN;             // 1048576

  const float* q    = (const float*)d_in[0];
  const float* s    = (const float*)d_in[1];
  const float* ts_w = (const float*)d_in[2];
  const float* ts_b = (const float*)d_in[3];
  const float* ts_g = (const float*)d_in[4];
  const float* ts_be= (const float*)d_in[5];
  const float* ts_m = (const float*)d_in[6];
  const float* ts_v = (const float*)d_in[7];
  const float* tq_w = (const float*)d_in[8];
  const float* tq_b = (const float*)d_in[9];
  const float* tq_g = (const float*)d_in[10];
  const float* tq_be= (const float*)d_in[11];
  const float* tq_m = (const float*)d_in[12];
  const float* tq_v = (const float*)d_in[13];
  const float* cv_w = (const float*)d_in[14];
  const float* cv_b = (const float*)d_in[15];
  const float* k1_w = (const float*)d_in[16];
  const float* k1_b = (const float*)d_in[17];
  const float* q1_w = (const float*)d_in[18];
  const float* q1_b = (const float*)d_in[19];
  const float* k2_w = (const float*)d_in[20];
  const float* k2_b = (const float*)d_in[21];
  const float* q2_w = (const float*)d_in[22];
  const float* q2_b = (const float*)d_in[23];
  const float* g1_w = (const float*)d_in[24];
  const float* g1_b = (const float*)d_in[25];
  const float* g2_w = (const float*)d_in[26];
  const float* g2_b = (const float*)d_in[27];
  const float* cc_w = (const float*)d_in[28];
  const float* cc_g = (const float*)d_in[29];
  const float* cc_be= (const float*)d_in[30];
  const float* cc_m = (const float*)d_in[31];
  const float* cc_v = (const float*)d_in[32];

  // ---- workspace carve ----
  char* wsp = (char*)d_ws;
  size_t off = 0;
  auto alloc = [&](size_t bytes) -> void* {
    void* p = wsp + off;
    off = (off + bytes + 255) & ~(size_t)255;
    return p;
  };
  const long nQS   = (long)B * Cin * N;
  __bf16* q_bf      = (__bf16*)alloc(nQS * 2);
  __bf16* s_bf      = (__bf16*)alloc(nQS * 2);
  __bf16* ts_w_bf   = (__bf16*)alloc((size_t)Ci * Cin * 9 * 2);
  __bf16* tq_w_bf   = (__bf16*)alloc((size_t)Ci * Cin * 9 * 2);
  __bf16* cc_w_bf   = (__bf16*)alloc((size_t)Ci * (2 * Ci) * 9 * 2);
  __bf16* cv_w_bf   = (__bf16*)alloc((size_t)Ci * Ci * 2);
  __bf16* k1_w_bf   = (__bf16*)alloc((size_t)(Ci / 2) * Ci * 2);
  __bf16* q1_w_bf   = (__bf16*)alloc((size_t)(Ci / 2) * Ci * 2);
  __bf16* k2_w_bf   = (__bf16*)alloc((size_t)(Ci / 2) * Ci * 2);
  __bf16* q2_w_bf   = (__bf16*)alloc((size_t)(Ci / 2) * Ci * 2);
  float*  s_stem_f  = (float*)alloc(BCN * 4);
  float*  q_stem_f  = (float*)alloc(BCN * 4);
  __bf16* s_stem_bf = (__bf16*)alloc(BCN * 2);
  __bf16* q_stem_bf = (__bf16*)alloc(BCN * 2);
  float*  v_s1      = (float*)alloc(BCN * 4);
  float*  v_q1      = (float*)alloc(BCN * 4);
  __bf16* v_s_bf    = (__bf16*)alloc(BCN * 2);   // [B][N,C]
  __bf16* v_q_bf    = (__bf16*)alloc(BCN * 2);   // [B][N,C]
  __bf16* key_bf    = (__bf16*)alloc(BCN * 2);   // [B][N,C]
  __bf16* qry_bf    = (__bf16*)alloc(BCN * 2);   // [B][C,N]
  float*  p_s       = (float*)alloc(BCN * 4);
  float*  q_s       = (float*)alloc(BCN * 4);
  __bf16* ecat_bf   = (__bf16*)alloc((size_t)B * 2 * Ci * N * 2); // [B][2Ci,N]
  float*  pooled_s  = (float*)alloc((size_t)B * Ci * 4);
  float*  pooled_q  = (float*)alloc((size_t)B * Ci * 4);
  float*  gate_s    = (float*)alloc((size_t)B * Ci * 4);
  float*  gate_q    = (float*)alloc((size_t)B * Ci * 4);

  float* out_cpam = (float*)d_out;
  float* out_Eq   = out_cpam + BCN;
  float* out_Es   = out_Eq + BCN;

  auto cvt = [&](const float* src, __bf16* dst, long n) {
    cvt_f32_bf16_kernel<<<dim3((unsigned)((n + 255) / 256)), dim3(256), 0, stream>>>(
        src, dst, (int)n);
  };
  cvt(q, q_bf, nQS);
  cvt(s, s_bf, nQS);
  cvt(ts_w, ts_w_bf, (long)Ci * Cin * 9);
  cvt(tq_w, tq_w_bf, (long)Ci * Cin * 9);
  cvt(cc_w, cc_w_bf, (long)Ci * (2 * Ci) * 9);
  cvt(cv_w, cv_w_bf, (long)Ci * Ci);
  cvt(k1_w, k1_w_bf, (long)(Ci / 2) * Ci);
  cvt(q1_w, q1_w_bf, (long)(Ci / 2) * Ci);
  cvt(k2_w, k2_w_bf, (long)(Ci / 2) * Ci);
  cvt(q2_w, q2_w_bf, (long)(Ci / 2) * Ci);

  const dim3 blk(256);
  // ---- stems: conv3x3 + BN + ReLU ----
  conv_gemm_kernel<3, 1, 64, 64><<<dim3(Ci / 16, N / 128, B), blk, 0, stream>>>(
      s_bf, (long)Cin * N, ts_w_bf, ts_b, ts_g, ts_be, ts_m, ts_v,
      s_stem_f, CN, N, 1, s_stem_bf, CN, N, 1,
      Ci, Cin, 1, 0);
  conv_gemm_kernel<3, 1, 64, 64><<<dim3(Ci / 16, N / 128, B), blk, 0, stream>>>(
      q_bf, (long)Cin * N, tq_w_bf, tq_b, tq_g, tq_be, tq_m, tq_v,
      q_stem_f, CN, N, 1, q_stem_bf, CN, N, 1,
      Ci, Cin, 1, 0);

  // ---- common value 1x1 ----
  conv_gemm_kernel<1, 0, 64, 64><<<dim3(Ci / 16, N / 128, B), blk, 0, stream>>>(
      s_stem_bf, CN, cv_w_bf, cv_b, nullptr, nullptr, nullptr, nullptr,
      v_s1, CN, N, 1, v_s_bf, CN, 1, Ci,
      Ci, Ci, 0, 0);
  conv_gemm_kernel<1, 0, 64, 64><<<dim3(Ci / 16, N / 128, B), blk, 0, stream>>>(
      q_stem_bf, CN, cv_w_bf, cv_b, nullptr, nullptr, nullptr, nullptr,
      v_q1, CN, N, 1, v_q_bf, CN, 1, Ci,
      Ci, Ci, 0, 0);

  // ---- key_cat = [k1(s); -k2(q)] packed as [B][N,128] bf16 ----
  conv_gemm_kernel<1, 0, 64, 64><<<dim3(Ci / 32, N / 128, B), blk, 0, stream>>>(
      s_stem_bf, CN, k1_w_bf, k1_b, nullptr, nullptr, nullptr, nullptr,
      nullptr, 0, 0, 0, key_bf, CN, 1, Ci,
      Ci / 2, Ci, 0, 0);
  conv_gemm_kernel<1, 0, 64, 64><<<dim3(Ci / 32, N / 128, B), blk, 0, stream>>>(
      q_stem_bf, CN, k2_w_bf, k2_b, nullptr, nullptr, nullptr, nullptr,
      nullptr, 0, 0, 0, key_bf + (Ci / 2), CN, 1, Ci,
      Ci / 2, Ci, 0, 1 /*negate*/);

  // ---- query_cat = [q1(s); q2(q)] packed as [B][128,N] bf16 ----
  conv_gemm_kernel<1, 0, 64, 64><<<dim3(Ci / 32, N / 128, B), blk, 0, stream>>>(
      s_stem_bf, CN, q1_w_bf, q1_b, nullptr, nullptr, nullptr, nullptr,
      nullptr, 0, 0, 0, qry_bf, CN, N, 1,
      Ci / 2, Ci, 0, 0);
  conv_gemm_kernel<1, 0, 64, 64><<<dim3(Ci / 32, N / 128, B), blk, 0, stream>>>(
      q_stem_bf, CN, q2_w_bf, q2_b, nullptr, nullptr, nullptr, nullptr,
      nullptr, 0, 0, 0, qry_bf + (long)(Ci / 2) * N, CN, N, 1,
      Ci / 2, Ci, 0, 0);

  // ---- flash attention ----
  attn_kernel<true, false><<<dim3(N / 64, B), blk, 0, stream>>>(
      key_bf, Ci, 1, CN,      // A (n,c) contiguous c  -> async/vector copy
      qry_bf, N, 1, CN,       // B (c,m) contiguous m  -> transpose path
      v_s_bf, CN, p_s, CN, N, Ci);
  attn_kernel<false, true><<<dim3(N / 64, B), blk, 0, stream>>>(
      qry_bf, 1, N, CN,       // A (n,c) contiguous n  -> transpose path
      key_bf, 1, Ci, CN,      // B (c,m) contiguous c  -> async/vector copy
      v_q_bf, CN, q_s, CN, N, Ci);

  // ---- channel gates ----
  pool_kernel<<<dim3(B * Ci), blk, 0, stream>>>(v_s1, pooled_s, N);
  pool_kernel<<<dim3(B * Ci), blk, 0, stream>>>(v_q1, pooled_q, N);
  gate_kernel<<<dim3(1), blk, 0, stream>>>(pooled_s, g1_w, g1_b, g2_w, g2_b,
                                           gate_s, B, Ci, Cr);
  gate_kernel<<<dim3(1), blk, 0, stream>>>(pooled_q, g1_w, g1_b, g2_w, g2_b,
                                           gate_q, B, Ci, Cr);

  // ---- E_q / E_s ----
  fuse_kernel<<<dim3((unsigned)((BCN + 255) / 256)), blk, 0, stream>>>(
      q_s, q_stem_f, gate_q, out_Eq, ecat_bf, Ci, N, 0, 2 * Ci, BCN);
  fuse_kernel<<<dim3((unsigned)((BCN + 255) / 256)), blk, 0, stream>>>(
      p_s, s_stem_f, gate_s, out_Es, ecat_bf, Ci, N, Ci, 2 * Ci, BCN);

  // ---- conv_cat ----
  conv_gemm_kernel<3, 1, 64, 64><<<dim3(Ci / 16, N / 128, B), blk, 0, stream>>>(
      ecat_bf, (long)2 * Ci * N, cc_w_bf, nullptr, cc_g, cc_be, cc_m, cc_v,
      out_cpam, CN, N, 1, nullptr, 0, 0, 0,
      Ci, 2 * Ci, 1, 0);
}
